// NodeChildSumTreeLSTMCell_56882546868972
// MI455X (gfx1250) — compile-verified, matrix operand order checked by reading, not run on hardware
//
#include <hip/hip_runtime.h>
#include <math.h>

typedef __attribute__((ext_vector_type(16))) _Float16 v16h;
typedef __attribute__((ext_vector_type(8)))  _Float16 v8h;
typedef __attribute__((ext_vector_type(8)))  float    v8f;

#define H128 128

// ---------------------------------------------------------------------------
// WMMA fragment helpers (CDNA5 wave32, 16x16x32 f16 -> f32)
// 16-bit A layout (ISA 7.12.2): lanes 0-15 hold M=0-15,
//   VGPR0..3 = K(koff+0..7), VGPR4..7 = K(koff+16..23), koff = 8*(lane>=16).
// B fragments use the same pattern with N in place of M because weights are
// stored transposed ([n][k] row-major).
// ---------------------------------------------------------------------------
__device__ __forceinline__ v16h load_frag(const _Float16* __restrict__ base,
                                          int ld, int lane, int k0) {
  const int r    = lane & 15;
  const int koff = (lane & 16) >> 1;            // 0 or 8
  const _Float16* p = base + r * ld + k0 + koff;
  v8h lo = *(const v8h*)(p);
  v8h hi = *(const v8h*)(p + 16);
  v16h a;
#pragma unroll
  for (int j = 0; j < 8; ++j) { a[j] = lo[j]; a[8 + j] = hi[j]; }
  return a;
}

__device__ __forceinline__ v8f wmma16(v16h a, v16h b, v8f c) {
  return __builtin_amdgcn_wmma_f32_16x16x32_f16(false, a, false, b,
                                                (short)0, c, false, false);
}

// C/D layout: VGPR v -> (M=v, N=lane) lanes 0-15, (M=v+8, N=lane-16) lanes 16-31.
__device__ __forceinline__ void store_d(float* dst, int ld, int lane, v8f d) {
  const int n  = lane & 15;
  const int mb = (lane & 16) >> 1;
#pragma unroll
  for (int v = 0; v < 8; ++v) dst[(mb + v) * ld + n] = d[v];
}

__device__ __forceinline__ v8f load_c(const float* src, int ld, int lane) {
  const int n  = lane & 15;
  const int mb = (lane & 16) >> 1;
  v8f d;
#pragma unroll
  for (int v = 0; v < 8; ++v) d[v] = src[(mb + v) * ld + n];
  return d;
}

__device__ __forceinline__ float sigm(float v) {
  return 1.0f / (1.0f + __expf(-v));
}

// ---------------------------------------------------------------------------
// Weight prep: f32 -> f16, transposed to [out_col][k].
//   WcatT [512][128] : rows 0-383 = W_iou cols, 384-511 = W_f cols
//   UiouT [384][128], UfT [128][128], bcat[512] = {b_iou, b_f}
// ---------------------------------------------------------------------------
__global__ void prep_weights(const float* __restrict__ W_iou,
                             const float* __restrict__ U_iou,
                             const float* __restrict__ b_iou,
                             const float* __restrict__ W_f,
                             const float* __restrict__ U_f,
                             const float* __restrict__ b_f,
                             _Float16* __restrict__ WcatT,
                             _Float16* __restrict__ UiouT,
                             _Float16* __restrict__ UfT,
                             float* __restrict__ bcat) {
  const int TOTAL = 512 * 128 + 384 * 128 + 128 * 128 + 512;
  for (int t = blockIdx.x * blockDim.x + threadIdx.x; t < TOTAL;
       t += gridDim.x * blockDim.x) {
    if (t < 512 * 128) {
      int o = t >> 7, k = t & 127;
      float v = (o < 384) ? W_iou[k * 384 + o] : W_f[k * 128 + (o - 384)];
      WcatT[t] = (_Float16)v;
    } else if (t < 512 * 128 + 384 * 128) {
      int u = t - 512 * 128;
      int o = u >> 7, k = u & 127;
      UiouT[u] = (_Float16)U_iou[k * 384 + o];
    } else if (t < 512 * 128 + 384 * 128 + 128 * 128) {
      int u = t - (512 * 128 + 384 * 128);
      int o = u >> 7, k = u & 127;
      UfT[u] = (_Float16)U_f[k * 128 + o];
    } else {
      int u = t - (512 * 128 + 384 * 128 + 128 * 128);
      bcat[u] = (u < 384) ? b_iou[u] : b_f[u - 384];
    }
  }
}

// ---------------------------------------------------------------------------
// Leaf kernel: 32 nodes / group, 384 threads (12 waves).
// Only iou (384 cols) is needed at leaves: c = sig(i)*tanh(u), h = sig(o)*tanh(c).
// Each wave: 2 N-tiles x 2 M-tiles, B fragment loaded once per (nt,k).
// ---------------------------------------------------------------------------
__global__ __launch_bounds__(384)
void tree_leaf_kernel(const float* __restrict__ x,
                      const _Float16* __restrict__ WcatT,   // use rows 0..383
                      const float* __restrict__ bcat,
                      float* __restrict__ h_out,
                      float* __restrict__ c_out,
                      int s, int nl) {
  __shared__ __align__(16) _Float16 s_x16[32 * 128];   //  8 KB
  __shared__ __align__(16) float    s_iou[32 * 384];   // 48 KB

  const int tid  = threadIdx.x;
  const int lane = tid & 31;
  const int wave = tid >> 5;               // 0..11
  const int g0   = blockIdx.x * 32;

  for (int idx = tid; idx < 32 * 128; idx += 384) {
    int i = idx >> 7, g = idx & 127;
    int node = g0 + i;
    float v = (node < nl) ? x[(size_t)(s + node) * H128 + g] : 0.0f;
    s_x16[idx] = (_Float16)v;
  }
  __syncthreads();

  v16h ax[2][4];
#pragma unroll
  for (int mt = 0; mt < 2; ++mt)
#pragma unroll
    for (int kk = 0; kk < 4; ++kk)
      ax[mt][kk] = load_frag(s_x16 + mt * 16 * 128, 128, lane, kk * 32);

#pragma unroll
  for (int p = 0; p < 2; ++p) {
    int n0 = (wave * 2 + p) * 16;                 // 0..368
    float bv = bcat[n0 + (lane & 15)];
    v8f acc0, acc1;
#pragma unroll
    for (int v = 0; v < 8; ++v) { acc0[v] = bv; acc1[v] = bv; }
    const _Float16* bt = WcatT + (size_t)n0 * 128;
#pragma unroll
    for (int kk = 0; kk < 4; ++kk) {
      v16h b = load_frag(bt, 128, lane, kk * 32);
      acc0 = wmma16(ax[0][kk], b, acc0);
      acc1 = wmma16(ax[1][kk], b, acc1);
    }
    store_d(&s_iou[n0], 384, lane, acc0);
    store_d(&s_iou[16 * 384 + n0], 384, lane, acc1);
  }
  __syncthreads();

  for (int idx = tid; idx < 32 * 128; idx += 384) {
    int i = idx >> 7, g = idx & 127;
    int node = g0 + i;
    if (node >= nl) continue;
    float ig = s_iou[i * 384 + g];
    float og = s_iou[i * 384 + 128 + g];
    float ug = s_iou[i * 384 + 256 + g];
    float c = sigm(ig) * tanhf(ug);
    float h = sigm(og) * tanhf(c);
    size_t o = (size_t)(s + node) * H128 + g;
    h_out[o] = h;
    c_out[o] = c;
  }
}

// ---------------------------------------------------------------------------
// Internal-level kernel: 32 nodes / group, 512 threads (16 waves).
//   GEMM1: x[32x128] @ Wcat[128x512]        (B reuse across 2 M-tiles)
//   GEMM2: h_tilda[32x128] @ Uiou[128x384] += node_iou
//   GEMM3: ch_h[128x128] @ Uf[128x128]      (B reuse across 4 M-tiles)
// ---------------------------------------------------------------------------
__global__ __launch_bounds__(512)
void tree_internal_kernel(const float* __restrict__ x,
                          const _Float16* __restrict__ WcatT,
                          const _Float16* __restrict__ UiouT,
                          const _Float16* __restrict__ UfT,
                          const float* __restrict__ bcat,
                          float* __restrict__ h_out,
                          float* __restrict__ c_out,
                          int s, int nl, int child_base) {
  __shared__ __align__(16) _Float16 s_x16[32 * 128];    //   8 KB
  __shared__ __align__(16) _Float16 s_h16[128 * 128];   //  32 KB
  __shared__ __align__(16) _Float16 s_ht16[32 * 128];   //   8 KB
  __shared__ __align__(16) float    s_iou[32 * 384];    //  48 KB
  __shared__ __align__(16) float    s_nf[32 * 128];     //  16 KB
  __shared__ __align__(16) float    s_f[128 * 128];     //  64 KB

  const int tid  = threadIdx.x;
  const int lane = tid & 31;
  const int wave = tid >> 5;               // 0..15
  const int g0   = blockIdx.x * 32;

  // ---- stage x tile ----
  for (int idx = tid; idx < 32 * 128; idx += 512) {
    int i = idx >> 7, g = idx & 127;
    int node = g0 + i;
    float v = (node < nl) ? x[(size_t)(s + node) * H128 + g] : 0.0f;
    s_x16[idx] = (_Float16)v;
  }
  // ---- stage child h + h_tilda ----
  for (int idx = tid; idx < 32 * 128; idx += 512) {
    int i = idx >> 7, g = idx & 127;
    int node = g0 + i;
    float sum = 0.0f;
    if (node < nl) {
      int cb = child_base + node * 4;
#pragma unroll
      for (int k = 0; k < 4; ++k) {
        float hv = h_out[(size_t)(cb + k) * H128 + g];
        s_h16[(i * 4 + k) * H128 + g] = (_Float16)hv;
        sum += hv;
      }
    } else {
#pragma unroll
      for (int k = 0; k < 4; ++k) s_h16[(i * 4 + k) * H128 + g] = (_Float16)0.0f;
    }
    s_ht16[idx] = (_Float16)sum;
  }
  __syncthreads();

  // ---- GEMM1: 32 N-tiles x 2 M-tiles; wave owns 2 N-tiles ----
  v16h ax[2][4];
#pragma unroll
  for (int mt = 0; mt < 2; ++mt)
#pragma unroll
    for (int kk = 0; kk < 4; ++kk)
      ax[mt][kk] = load_frag(s_x16 + mt * 16 * 128, 128, lane, kk * 32);

#pragma unroll
  for (int p = 0; p < 2; ++p) {
    int n0 = (wave * 2 + p) * 16;                 // 0..496
    float bv = bcat[n0 + (lane & 15)];
    v8f acc0, acc1;
#pragma unroll
    for (int v = 0; v < 8; ++v) { acc0[v] = bv; acc1[v] = bv; }
    const _Float16* bt = WcatT + (size_t)n0 * 128;
#pragma unroll
    for (int kk = 0; kk < 4; ++kk) {
      v16h b = load_frag(bt, 128, lane, kk * 32);
      acc0 = wmma16(ax[0][kk], b, acc0);
      acc1 = wmma16(ax[1][kk], b, acc1);
    }
    if (n0 < 384) {
      store_d(&s_iou[n0], 384, lane, acc0);
      store_d(&s_iou[16 * 384 + n0], 384, lane, acc1);
    } else {
      store_d(&s_nf[n0 - 384], 128, lane, acc0);
      store_d(&s_nf[16 * 128 + (n0 - 384)], 128, lane, acc1);
    }
  }

  // ---- GEMM2: 24 N-tiles x 2 M-tiles = 48 tiles, 3 per wave ----
  v16h ah[2][4];
#pragma unroll
  for (int mt = 0; mt < 2; ++mt)
#pragma unroll
    for (int kk = 0; kk < 4; ++kk)
      ah[mt][kk] = load_frag(s_ht16 + mt * 16 * 128, 128, lane, kk * 32);
  __syncthreads();                 // all GEMM1 s_iou stores visible
#pragma unroll
  for (int q = 0; q < 3; ++q) {
    int t  = wave * 3 + q;                        // 0..47
    int nt = t >> 1, mt = t & 1;
    int n0 = nt * 16;
    float* cbase = &s_iou[mt * 16 * 384 + n0];
    v8f acc = load_c(cbase, 384, lane);
    const _Float16* bt = UiouT + (size_t)n0 * 128;
#pragma unroll
    for (int kk = 0; kk < 4; ++kk) {
      v16h b = load_frag(bt, 128, lane, kk * 32);
      acc = wmma16(ah[mt][kk], b, acc);
    }
    store_d(cbase, 384, lane, acc);
  }

  // ---- GEMM3: ch_h[128x128] @ Uf -> s_f. 8 N x 8 M tiles; wave owns one
  //      N column over 4 M-tiles; B loaded once per (nt,k), reused 4x. ----
  {
    int nt  = wave & 7;
    int mtb = (wave >> 3) * 4;
    int n0  = nt * 16;
    v8f acc[4] = {};
    const _Float16* bt = UfT + (size_t)n0 * 128;
#pragma unroll
    for (int kk = 0; kk < 4; ++kk) {
      v16h b = load_frag(bt, 128, lane, kk * 32);
#pragma unroll
      for (int q = 0; q < 4; ++q) {
        v16h a = load_frag(s_h16 + (size_t)(mtb + q) * 16 * 128, 128, lane, kk * 32);
        acc[q] = wmma16(a, b, acc[q]);
      }
    }
#pragma unroll
    for (int q = 0; q < 4; ++q)
      store_d(&s_f[(size_t)(mtb + q) * 16 * 128 + n0], 128, lane, acc[q]);
  }
  __syncthreads();

  // ---- gates + cell/hidden update ----
  for (int idx = tid; idx < 32 * 128; idx += 512) {
    int i = idx >> 7, g = idx & 127;
    int node = g0 + i;
    if (node >= nl) continue;
    int cb = child_base + node * 4;
    float nf = s_nf[i * 128 + g];
    float csum = 0.0f;
#pragma unroll
    for (int k = 0; k < 4; ++k) {
      float fpre = s_f[(i * 4 + k) * 128 + g] + nf;
      float cc   = c_out[(size_t)(cb + k) * H128 + g];
      csum += sigm(fpre) * cc;
    }
    float ig = s_iou[i * 384 + g];
    float og = s_iou[i * 384 + 128 + g];
    float ug = s_iou[i * 384 + 256 + g];
    float c = sigm(ig) * tanhf(ug) + csum;
    float h = sigm(og) * tanhf(c);
    size_t o = (size_t)(s + node) * H128 + g;
    h_out[o] = h;
    c_out[o] = c;
  }
}

// ---------------------------------------------------------------------------
extern "C" void kernel_launch(void* const* d_in, const int* in_sizes, int n_in,
                              void* d_out, int out_size, void* d_ws, size_t ws_size,
                              hipStream_t stream) {
  const float* x     = (const float*)d_in[0];
  const float* W_iou = (const float*)d_in[1];
  const float* U_iou = (const float*)d_in[2];
  const float* b_iou = (const float*)d_in[3];
  const float* W_f   = (const float*)d_in[4];
  const float* U_f   = (const float*)d_in[5];
  const float* b_f   = (const float*)d_in[6];
  // d_in[7]=depth, d_in[8]=branching: device scalars; fixed by reference.

  const int branching = 4;
  const long long N = (long long)in_sizes[0] / H128;

  int off[34];
  off[0] = 0;
  int depth = 0;
  long long cnt = 1, tot = 0;
  while (tot < N && depth < 32) {
    tot += cnt;
    depth++;
    off[depth] = (int)tot;
    cnt *= branching;
  }

  char* ws = (char*)d_ws;
  _Float16* WcatT = (_Float16*)(ws);                            // 131072 B
  _Float16* UiouT = (_Float16*)(ws + 131072);                   //  98304 B
  _Float16* UfT   = (_Float16*)(ws + 131072 + 98304);           //  32768 B
  float*    bcat  = (float*)   (ws + 131072 + 98304 + 32768);   //   2048 B

  float* h_out = (float*)d_out;
  float* c_out = h_out + (size_t)N * H128;

  const int PREP_TOTAL = 512 * 128 + 384 * 128 + 128 * 128 + 512;
  prep_weights<<<(PREP_TOTAL + 255) / 256, 256, 0, stream>>>(
      W_iou, U_iou, b_iou, W_f, U_f, b_f, WcatT, UiouT, UfT, bcat);

  for (int l = depth - 1; l >= 0; --l) {
    int s  = off[l];
    int e  = off[l + 1];
    int nl = e - s;
    int groups = (nl + 31) / 32;
    if (l == depth - 1) {
      tree_leaf_kernel<<<groups, 384, 0, stream>>>(
          x, WcatT, bcat, h_out, c_out, s, nl);
    } else {
      tree_internal_kernel<<<groups, 512, 0, stream>>>(
          x, WcatT, UiouT, UfT, bcat, h_out, c_out, s, nl, e);
    }
  }
}
